// HAwareGateElman_47768626266225
// MI455X (gfx1250) — compile-verified, matrix operand order checked by reading, not run on hardware
//
#include <hip/hip_runtime.h>
#include <cstdint>
#include <cstddef>

#define D_DIM 1024
#define T_DIM 2048
#define B_DIM 8
#define M_DIM (B_DIM * T_DIM)   // 16384

typedef __attribute__((ext_vector_type(16))) __bf16        v16bf;
typedef __attribute__((ext_vector_type(8)))  float         v8f;
typedef __attribute__((ext_vector_type(8)))  unsigned int  v8u;

__device__ __forceinline__ unsigned short f2bf(float f) {
  union { float f; unsigned u; } x; x.f = f;
  unsigned r = x.u + 0x7FFFu + ((x.u >> 16) & 1u);   // RNE
  return (unsigned short)(r >> 16);
}

__device__ __forceinline__ float silu_f(float x) {
  return x / (1.f + __expf(-x));
}

// ---- CDNA5 async global->LDS copy (ASYNCcnt-tracked, no VGPR round trip) ----
// Generic LDS addresses keep the LDS offset in addr[31:0] (ISA 10.2 aperture
// rules), so the low 32 bits of a __shared__ pointer are a valid VDST operand.
__device__ __forceinline__ unsigned lds_addr_of(const void* p) {
  return (unsigned)(uintptr_t)p;
}
__device__ __forceinline__ void async_copy_b128(unsigned lds_addr, const void* gsrc) {
  asm volatile("global_load_async_to_lds_b128 %0, %1, off"
               :: "v"(lds_addr), "v"((unsigned long long)(uintptr_t)gsrc)
               : "memory");
}
__device__ __forceinline__ void wait_async0() {
  asm volatile("s_wait_asynccnt 0x0" ::: "memory");
}

// ---------------------------------------------------------------------------
// Tiled bf16 WMMA GEMM: C(M,N) = A(M,K) @ B(N,K)^T
// Block tile 128x128, K-tile 32, 256 threads = 8 waves, each wave 2x4 tiles.
// Double-buffered LDS; B (and bf16 A) tiles staged with async-to-LDS copies.
// EPI 0: silu-split (cols < Dsplit -> silu -> bf16 xp; else -> f32 z)
// EPI 1: + bias -> f32
// EPI 2: plain -> f32
// ---------------------------------------------------------------------------
#define BM 128
#define BN 128
#define BK 32
#define LDS_STRIDE 48   // bf16 elems per LDS row (pad; rows stay 16B aligned)

template<int EPI, bool A_F32>
__global__ __launch_bounds__(256)
void gemm_bf16_wmma(const void* __restrict__ Aptr,
                    const unsigned short* __restrict__ Bw,
                    int N, int K,
                    const float* __restrict__ bias,
                    unsigned short* __restrict__ out_bf,
                    float* __restrict__ out_f32,
                    int Dsplit)
{
  __shared__ unsigned short As[2][BM * LDS_STRIDE];
  __shared__ unsigned short Bs[2][BN * LDS_STRIDE];

  const int tid  = threadIdx.x;
  const int lane = tid & 31;
  const int wave = tid >> 5;     // 0..7
  const int wm   = wave >> 1;    // 0..3 -> 32-row slab
  const int wn   = wave & 1;     // 0..1 -> 64-col slab
  const int bm   = blockIdx.y * BM;
  const int bn   = blockIdx.x * BN;

  const int ldr = tid >> 1;          // 0..127 staged tile row
  const int ldc = (tid & 1) * 16;    // 0 or 16

  v8f acc[2][4] = {};

  const int lrow = lane & 15;
  const int lk   = (lane >> 4) * 16;

  auto stage = [&](int k0, int buf) {
    // B tile: async global->LDS (weights already bf16, row-major [N][K])
    {
      const unsigned short* src = Bw + (size_t)(bn + ldr) * K + k0 + ldc;
      unsigned l = lds_addr_of(&Bs[buf][ldr * LDS_STRIDE + ldc]);
      async_copy_b128(l,      src);
      async_copy_b128(l + 16, src + 8);
    }
    if (A_F32) {
      // fp32 source: convert in-register, synchronous ds stores
      const float* src = (const float*)Aptr + (size_t)(bm + ldr) * K + k0 + ldc;
      unsigned int* dst = (unsigned int*)&As[buf][ldr * LDS_STRIDE + ldc];
      #pragma unroll
      for (int i = 0; i < 8; ++i)
        dst[i] = (unsigned)f2bf(src[2*i]) | ((unsigned)f2bf(src[2*i+1]) << 16);
    } else {
      const unsigned short* src =
        (const unsigned short*)Aptr + (size_t)(bm + ldr) * K + k0 + ldc;
      unsigned l = lds_addr_of(&As[buf][ldr * LDS_STRIDE + ldc]);
      async_copy_b128(l,      src);
      async_copy_b128(l + 16, src + 8);
    }
  };

  stage(0, 0);
  wait_async0();
  __syncthreads();

  int cur = 0;
  for (int k0 = 0; k0 < K; k0 += BK) {
    if (k0 + BK < K) stage(k0 + BK, cur ^ 1);   // fill next buffer while computing

    v16bf afrag[2], bfrag[4];
    #pragma unroll
    for (int mt = 0; mt < 2; ++mt)
      afrag[mt] = __builtin_bit_cast(
          v16bf, *(const v8u*)&As[cur][(wm*32 + mt*16 + lrow) * LDS_STRIDE + lk]);
    #pragma unroll
    for (int nt = 0; nt < 4; ++nt)
      bfrag[nt] = __builtin_bit_cast(
          v16bf, *(const v8u*)&Bs[cur][(wn*64 + nt*16 + lrow) * LDS_STRIDE + lk]);

    #pragma unroll
    for (int mt = 0; mt < 2; ++mt)
      #pragma unroll
      for (int nt = 0; nt < 4; ++nt)
        acc[mt][nt] = __builtin_amdgcn_wmma_f32_16x16x32_bf16(
            false, afrag[mt], false, bfrag[nt], (short)0, acc[mt][nt], false, false);

    wait_async0();       // next tile landed in LDS
    __syncthreads();     // everyone done reading cur
    cur ^= 1;
  }

  // ---- epilogue: VGPR r holds rows r (lanes 0-15) and r+8 (lanes 16-31) ----
  #pragma unroll
  for (int mt = 0; mt < 2; ++mt) {
    #pragma unroll
    for (int nt = 0; nt < 4; ++nt) {
      #pragma unroll
      for (int r = 0; r < 8; ++r) {
        int grow = bm + wm*32 + mt*16 + r + 8*(lane >> 4);
        int gcol = bn + wn*64 + nt*16 + (lane & 15);
        float v = acc[mt][nt][r];
        if (EPI == 0) {
          if (gcol < Dsplit) out_bf[(size_t)grow * Dsplit + gcol] = f2bf(silu_f(v));
          else               out_f32[(size_t)grow * Dsplit + (gcol - Dsplit)] = v;
        } else if (EPI == 1) {
          out_f32[(size_t)grow * N + gcol] = v + bias[gcol];
        } else {
          out_f32[(size_t)grow * N + gcol] = v;
        }
      }
    }
  }
}

// ---------------------------------------------------------------------------
// Persistent recurrent scan: 16 workgroups x 4 waves = 64 waves, one 16-col
// tile each.  W_h_sn slice pinned in LDS; h staged per-block into LDS with
// async copies each step (shared by all 4 waves); device barrier per step.
// LDS: 4*32.5KB (W) + 32.5KB (h stage) ~ 166KB of 320KB.
// ---------------------------------------------------------------------------
#define RT_BLOCKS 16
#define RT_WAVES  4
#define W_LDS_STRIDE 1040   // 1024 + pad, rows stay 16B/32B aligned (2080 B)

__global__ __launch_bounds__(128)
void recurrent_scan(const unsigned short* __restrict__ Wh_bf,   // [1024][1024]
                    const float* __restrict__ xw,               // (B,T,D) f32
                    const float* __restrict__ z,                // (B,T,D) f32
                    unsigned short* __restrict__ h_bf,          // [2][16][1024]
                    unsigned short* __restrict__ cell_bf,       // (B,T,D) bf16
                    float* __restrict__ h_final,                // (B,D) f32
                    unsigned int* __restrict__ bar)
{
  extern __shared__ unsigned short Wl[];
  const int tid  = threadIdx.x;
  const int lane = tid & 31;
  const int wave = tid >> 5;
  const int gw   = blockIdx.x * RT_WAVES + wave;   // 0..63 output col tile
  unsigned short* myW = Wl + (size_t)wave * 16 * W_LDS_STRIDE;
  unsigned short* hs  = Wl + (size_t)RT_WAVES * 16 * W_LDS_STRIDE;  // h stage

  // preload this wave's 16 rows of W_h_sn into LDS via async copies (32 KB)
  {
    int prow  = lane & 15;
    int phalf = (lane >> 4) * 512;
    const unsigned short* src = Wh_bf + (size_t)(gw*16 + prow) * D_DIM + phalf;
    unsigned l = lds_addr_of(myW + prow * W_LDS_STRIDE + phalf);
    #pragma unroll 8
    for (int i = 0; i < 512; i += 8)
      async_copy_b128(l + 2*i, src + i);
  }
  wait_async0();
  __syncthreads();

  const int lrow = lane & 15;
  const int lk   = (lane >> 4) * 16;
  const int col  = gw * 16 + lrow;
  const int hrow = (lane >> 4) * 8;

  const int srow = tid >> 3;          // h-stage: 128 threads cover 16 rows
  const int sseg = (tid & 7) * 128;   // 128 bf16 (256 B) per thread

  for (int t = 0; t < T_DIM; ++t) {
    const unsigned short* hsrc = h_bf + (size_t)(t & 1)       * 16 * D_DIM;
    unsigned short*       hdst = h_bf + (size_t)((t + 1) & 1) * 16 * D_DIM;

    // ---- stage h (32 KB, shared by all 4 waves) into LDS ----
    {
      const unsigned short* src = hsrc + (size_t)srow * D_DIM + sseg;
      unsigned l = lds_addr_of(hs + srow * W_LDS_STRIDE + sseg);
      #pragma unroll
      for (int i = 0; i < 128; i += 8)
        async_copy_b128(l + 2*i, src + i);
    }
    wait_async0();
    __syncthreads();

    v8f acc = {};
    #pragma unroll 8
    for (int kk = 0; kk < 32; ++kk) {
      int kb = kk * 32 + lk;
      v16bf a = __builtin_bit_cast(v16bf, *(const v8u*)(hs  + lrow * W_LDS_STRIDE + kb));
      v16bf b = __builtin_bit_cast(v16bf, *(const v8u*)(myW + lrow * W_LDS_STRIDE + kb));
      acc = __builtin_amdgcn_wmma_f32_16x16x32_bf16(
          false, a, false, b, (short)0, acc, false, false);
    }

    #pragma unroll
    for (int r = 0; r < 8; ++r) {
      int row = hrow + r;                 // lanes 16-31 cover pad rows 8..15
      if (row < B_DIM) {
        size_t idx = ((size_t)row * T_DIM + t) * D_DIM + col;
        float hn = tanhf(xw[idx] + acc[r]);
        float g  = z[idx] + hn;
        float o  = hn * (g / (1.f + __expf(-g)));   // hn * silu(z + hn)
        cell_bf[idx] = f2bf(o);
        hdst[(size_t)row * D_DIM + col] = f2bf(hn);
        if (t == T_DIM - 1) h_final[(size_t)row * D_DIM + col] = hn;
        if (t + 16 < T_DIM) {             // warm L2/L0 for future timesteps
          __builtin_prefetch(xw + idx + (size_t)16 * D_DIM, 0, 0);
          __builtin_prefetch(z  + idx + (size_t)16 * D_DIM, 0, 0);
        }
      }
    }

    // device-wide step barrier (monotonic counter, reset by init_state)
    __syncthreads();
    if (tid == 0) {
      __threadfence();
      atomicAdd(bar, 1u);
      unsigned int target = (unsigned int)RT_BLOCKS * (unsigned int)(t + 1);
      while (__hip_atomic_load(bar, __ATOMIC_ACQUIRE, __HIP_MEMORY_SCOPE_AGENT) < target) {
        __builtin_amdgcn_s_sleep(1);
      }
    }
    __syncthreads();
  }
}

// ---------------------------------------------------------------------------
// Small helper kernels (power iteration, conversions, init)
// ---------------------------------------------------------------------------
__global__ void vec_scale_norm(const float* __restrict__ src, float* __restrict__ dst,
                               int n, float eps)
{
  __shared__ float red[256];
  int tid = threadIdx.x;
  float s = 0.f;
  for (int i = tid; i < n; i += 256) { float v = src[i]; s += v * v; }
  red[tid] = s; __syncthreads();
  for (int off = 128; off > 0; off >>= 1) {
    if (tid < off) red[tid] += red[tid + off];
    __syncthreads();
  }
  float inv = 1.f / (sqrtf(red[0]) + eps);
  for (int i = tid; i < n; i += 256) dst[i] = src[i] * inv;
}

__global__ void matvec(const float* __restrict__ W, const float* __restrict__ x,
                       float* __restrict__ y, int rows, int cols, int transpose)
{
  int i = blockIdx.x * blockDim.x + threadIdx.x;
  if (!transpose) {
    if (i < rows) {
      float s = 0.f;
      for (int j = 0; j < cols; ++j) s += W[(size_t)i * cols + j] * x[j];
      y[i] = s;
    }
  } else {
    if (i < cols) {
      float s = 0.f;
      for (int j = 0; j < rows; ++j) s += W[(size_t)j * cols + i] * x[j];
      y[i] = s;
    }
  }
}

__global__ void dot_scale(const float* __restrict__ u, const float* __restrict__ w,
                          float* __restrict__ scale, int n)
{
  __shared__ float red[256];
  int tid = threadIdx.x;
  float s = 0.f;
  for (int i = tid; i < n; i += 256) s += u[i] * w[i];
  red[tid] = s; __syncthreads();
  for (int off = 128; off > 0; off >>= 1) {
    if (tid < off) red[tid] += red[tid + off];
    __syncthreads();
  }
  if (tid == 0) *scale = 0.99f / (fabsf(red[0]) + 1e-8f);
}

__global__ void convert_bf16(const float* __restrict__ src, unsigned short* __restrict__ dst,
                             const float* __restrict__ scale, size_t n)
{
  size_t i = (size_t)blockIdx.x * blockDim.x + threadIdx.x;
  if (i < n) {
    float s = scale ? *scale : 1.f;
    dst[i] = f2bf(src[i] * s);
  }
}

__global__ void init_state(const float* __restrict__ h0, unsigned short* __restrict__ h_bf,
                           unsigned int* __restrict__ bar)
{
  int i = blockIdx.x * blockDim.x + threadIdx.x;   // 2*16*1024 threads
  if (i == 0) *bar = 0u;
  int buf = i >> 14;
  int row = (i >> 10) & 15;
  int col = i & 1023;
  unsigned short v = 0;
  if (buf == 0 && row < B_DIM) v = f2bf(h0[row * D_DIM + col]);
  h_bf[i] = v;
}

// ---------------------------------------------------------------------------
extern "C" void kernel_launch(void* const* d_in, const int* in_sizes, int n_in,
                              void* d_out, int out_size, void* d_ws, size_t ws_size,
                              hipStream_t stream)
{
  const float* x     = (const float*)d_in[0];
  const float* h0    = (const float*)d_in[1];
  const float* u0    = (const float*)d_in[2];
  const float* W_in  = (const float*)d_in[3];
  const float* W_x   = (const float*)d_in[4];
  const float* W_h   = (const float*)d_in[5];
  const float* b     = (const float*)d_in[6];
  const float* W_out = (const float*)d_in[7];

  char* ws = (char*)d_ws;
  size_t off = 0;
  auto carve = [&](size_t bytes) -> void* {
    void* p = ws + off;
    off += (bytes + 255) & ~(size_t)255;
    return p;
  };

  float* u          = (float*)carve(D_DIM * 4);
  float* v          = (float*)carve(D_DIM * 4);
  float* wv         = (float*)carve(D_DIM * 4);
  float* scale      = (float*)carve(256);
  unsigned int* bar = (unsigned int*)carve(256);
  unsigned short* Winb = (unsigned short*)carve((size_t)2 * D_DIM * D_DIM * 2);
  unsigned short* Wxb  = (unsigned short*)carve((size_t)D_DIM * D_DIM * 2);
  unsigned short* Whb  = (unsigned short*)carve((size_t)D_DIM * D_DIM * 2);
  unsigned short* Wob  = (unsigned short*)carve((size_t)D_DIM * D_DIM * 2);
  unsigned short* xp   = (unsigned short*)carve((size_t)M_DIM * D_DIM * 2);
  float* zbuf          = (float*)carve((size_t)M_DIM * D_DIM * 4);
  float* xw            = (float*)carve((size_t)M_DIM * D_DIM * 4);
  unsigned short* cell = (unsigned short*)carve((size_t)M_DIM * D_DIM * 2);
  unsigned short* h_bf = (unsigned short*)carve((size_t)2 * 16 * D_DIM * 2);

  float* out   = (float*)d_out;
  float* h_fin = out + (size_t)M_DIM * D_DIM;

  // ---- spectral-norm power iteration (tiny, matches reference order) ----
  vec_scale_norm<<<1, 256, 0, stream>>>(u0, u, D_DIM, 0.f);
  for (int it = 0; it < 3; ++it) {
    matvec<<<4, 256, 0, stream>>>(W_h, u, v, D_DIM, D_DIM, 1);   // v = Wh^T u
    vec_scale_norm<<<1, 256, 0, stream>>>(v, v, D_DIM, 1e-8f);
    matvec<<<4, 256, 0, stream>>>(W_h, v, u, D_DIM, D_DIM, 0);   // u = Wh v
    vec_scale_norm<<<1, 256, 0, stream>>>(u, u, D_DIM, 1e-8f);
  }
  matvec<<<4, 256, 0, stream>>>(W_h, v, wv, D_DIM, D_DIM, 0);    // wv = Wh v
  dot_scale<<<1, 256, 0, stream>>>(u, wv, scale, D_DIM);         // 0.99/(|u.wv|+eps)

  // ---- weight conversions to bf16 (Wh gets the spectral scale folded in) ----
  {
    size_t n2 = (size_t)2 * D_DIM * D_DIM;
    size_t n1 = (size_t)D_DIM * D_DIM;
    convert_bf16<<<(unsigned)((n2 + 255) / 256), 256, 0, stream>>>(W_in,  Winb, nullptr, n2);
    convert_bf16<<<(unsigned)((n1 + 255) / 256), 256, 0, stream>>>(W_x,   Wxb,  nullptr, n1);
    convert_bf16<<<(unsigned)((n1 + 255) / 256), 256, 0, stream>>>(W_h,   Whb,  scale,   n1);
    convert_bf16<<<(unsigned)((n1 + 255) / 256), 256, 0, stream>>>(W_out, Wob,  nullptr, n1);
  }

  // ---- GEMM 1: xz = x @ W_in^T, silu-split epilogue ----
  gemm_bf16_wmma<0, true><<<dim3((2 * D_DIM) / BN, M_DIM / BM), 256, 0, stream>>>(
      x, Winb, 2 * D_DIM, D_DIM, nullptr, xp, zbuf, D_DIM);

  // ---- GEMM 2: xw = x_proj @ W_x^T + b ----
  gemm_bf16_wmma<1, false><<<dim3(D_DIM / BN, M_DIM / BM), 256, 0, stream>>>(
      xp, Wxb, D_DIM, D_DIM, b, nullptr, xw, 0);

  // ---- recurrent scan ----
  init_state<<<(2 * 16 * D_DIM) / 256, 256, 0, stream>>>(h0, h_bf, bar);
  size_t lds_bytes = (size_t)(RT_WAVES + 1) * 16 * W_LDS_STRIDE * 2;   // ~166 KB/WGP
  recurrent_scan<<<RT_BLOCKS, RT_WAVES * 32, lds_bytes, stream>>>(
      Whb, xw, zbuf, h_bf, cell, h_fin, bar);

  // ---- GEMM 3: output = cell_out @ W_out^T ----
  gemm_bf16_wmma<2, false><<<dim3(D_DIM / BN, M_DIM / BM), 256, 0, stream>>>(
      cell, Wob, D_DIM, D_DIM, nullptr, nullptr, out, 0);

  (void)in_sizes; (void)n_in; (void)out_size; (void)ws_size;
}
